// GraphConvMapperBlock_27152783245687
// MI455X (gfx1250) — matrix operand order repaired and checked by reading, not compiled, not run on hardware
//
#include <hip/hip_runtime.h>
#include <hip/hip_bf16.h>

// ---------------------------------------------------------------------------
// Types for CDNA5 WMMA (wave32): A/B = 16 bf16 (8 VGPRs), C/D = 8 f32.
// ---------------------------------------------------------------------------
typedef __bf16 bf16_t;
typedef __attribute__((ext_vector_type(16))) bf16_t v16bf;
typedef __attribute__((ext_vector_type(8)))  bf16_t v8bf;
typedef __attribute__((ext_vector_type(8)))  float  v8f;

#define DFEAT 128
#define LN_EPS 1e-5f

// A-fragment (16x32 bf16, M x K) from LDS, row-major with padded stride.
// Lane l: m = l&15, hi = l>>4. elements 0..7 = K kk*32 + hi*8 .. +7,
// elements 8..15 = K kk*32 + 16 + hi*8 .. +7  (per ISA 16-bit A layout).
__device__ __forceinline__ v16bf load_a_frag(const bf16_t* A, int stride,
                                             int kk, int m, int hi) {
  union { v16bf v; v8bf h[2]; } u;
  const bf16_t* p = A + m * stride + kk * 32 + hi * 8;
  u.h[0] = *(const v8bf*)p;        // 16B ds_load
  u.h[1] = *(const v8bf*)(p + 16); // 16B ds_load
  return u.v;
}

// B-fragment from pre-swizzled weights: 32 contiguous bytes per lane.
__device__ __forceinline__ v16bf load_b_frag(const bf16_t* W, int nt, int KK,
                                             int kk, int lane) {
  return *(const v16bf*)&W[(size_t)(((nt * KK + kk) * 32 + lane)) * 16];
}

__device__ __forceinline__ v8f wmma_bf16(v16bf a, v16bf b, v8f c) {
  return __builtin_amdgcn_wmma_f32_16x16x32_bf16(false, a, false, b,
                                                 (short)0, c, false, false);
}

__device__ __forceinline__ float silu(float x) {
  return x / (1.0f + __expf(-x));
}

__device__ __forceinline__ void atomic_add_f32(float* p, float v) {
  unsafeAtomicAdd(p, v); // native global_atomic_add_f32 (agg is L2-resident)
}

// ---------------------------------------------------------------------------
// Prep kernels
// ---------------------------------------------------------------------------
__global__ void zero_f32(float* __restrict__ p, int n) {
  int t = blockIdx.x * 256 + threadIdx.x;
  if (t < n) p[t] = 0.0f;
}

// Swizzle row-major fp32 weight W[K][128] into bf16 B-fragment order:
// dst[((nt*KK + kk)*32 + lane)*16 + j] = W[kk*32 + (j>>3)*16 + (lane>>4)*8 + (j&7)][nt*16 + (lane&15)]
__global__ void swizzle_w(const float* __restrict__ w, bf16_t* __restrict__ out,
                          int K) {
  int tid = blockIdx.x * 256 + threadIdx.x;
  int total = K * DFEAT;
  if (tid >= total) return;
  int j    = tid & 15;
  int lane = (tid >> 4) & 31;
  int f    = tid >> 9;
  int KK   = K >> 5;
  int nt   = f / KK;
  int kk   = f - nt * KK;
  int n = nt * 16 + (lane & 15);
  int k = kk * 32 + ((j >> 3) << 4) + ((lane >> 4) << 3) + (j & 7);
  out[tid] = (bf16_t)w[k * DFEAT + n];
}

// ---------------------------------------------------------------------------
// Edge MLP: 16 edges per block, 128 threads (4 waves, 2 N-tiles each).
// A = concat([x_dst[dst], x_src[src], edge_attr])  (K = 384)
// ---------------------------------------------------------------------------
#define EA_STRIDE 392   // 384 + 8 pad: row = 784B = 49*16B, banks advance by 4
#define H1_STRIDE 136   // 128 + 8 pad: row = 272B
#define H2_STRIDE 132   // fp32, row = 528B

__global__ __launch_bounds__(128)
void edge_kernel(const float* __restrict__ x_src, const float* __restrict__ x_dst,
                 const float* __restrict__ edge_attr, const int* __restrict__ eidx,
                 int n_edge,
                 const bf16_t* __restrict__ w1s, const float* __restrict__ b1,
                 const bf16_t* __restrict__ w2s, const float* __restrict__ b2,
                 const float* __restrict__ g, const float* __restrict__ beta,
                 float* __restrict__ out_edges, float* __restrict__ agg) {
  __shared__ __align__(16) bf16_t Asta[16 * EA_STRIDE];
  __shared__ __align__(16) bf16_t H1[16 * H1_STRIDE];
  __shared__ __align__(16) float  H2[16 * H2_STRIDE];
  __shared__ float part[16 * 8 * 2];
  __shared__ float stats[16 * 2];
  __shared__ int   sidx[16], didx[16];

  const int t  = threadIdx.x;
  const int e0 = blockIdx.x * 16;

  if (t < 16) {
    sidx[t] = eidx[e0 + t];           // edge_index[0] = src
    didx[t] = eidx[n_edge + e0 + t];  // edge_index[1] = dst
  }
  __syncthreads();

  // Stage A tile (16 x 384) as bf16: [x_i | x_j | edge_attr]
  {
    const int c = t;  // column 0..127, coalesced across threads
#pragma unroll 4
    for (int m = 0; m < 16; ++m) {
      Asta[m * EA_STRIDE + c]       = (bf16_t)x_dst[(size_t)didx[m] * DFEAT + c];
      Asta[m * EA_STRIDE + 128 + c] = (bf16_t)x_src[(size_t)sidx[m] * DFEAT + c];
      Asta[m * EA_STRIDE + 256 + c] = (bf16_t)edge_attr[(size_t)(e0 + m) * DFEAT + c];
    }
  }
  __syncthreads();

  const int lane = t & 31;
  const int wave = t >> 5;
  const int m16  = lane & 15;
  const int hi   = lane >> 4;
  const int nt0  = wave * 2;

  // ---- GEMM1: [16x384] x [384x128], K-steps of 32, C init = b1 ----
  v8f acc0, acc1;
  {
    float bv0 = b1[nt0 * 16 + m16];
    float bv1 = b1[(nt0 + 1) * 16 + m16];
#pragma unroll
    for (int i = 0; i < 8; ++i) { acc0[i] = bv0; acc1[i] = bv1; }
  }
#pragma unroll
  for (int kk = 0; kk < 12; ++kk) {
    v16bf a  = load_a_frag(Asta, EA_STRIDE, kk, m16, hi);
    v16bf bA = load_b_frag(w1s, nt0,     12, kk, lane);
    v16bf bB = load_b_frag(w1s, nt0 + 1, 12, kk, lane);
    acc0 = wmma_bf16(a, bA, acc0);
    acc1 = wmma_bf16(a, bB, acc1);
  }

  // SiLU -> bf16 restage (C layout: VGPR r -> row r + hi*8, col = tile*16+m16)
#pragma unroll
  for (int r = 0; r < 8; ++r) {
    int row = r + hi * 8;
    H1[row * H1_STRIDE + nt0 * 16 + m16]       = (bf16_t)silu(acc0[r]);
    H1[row * H1_STRIDE + (nt0 + 1) * 16 + m16] = (bf16_t)silu(acc1[r]);
  }
  __syncthreads();

  // ---- GEMM2: [16x128] x [128x128], C init = b2 ----
  v8f d0, d1;
  {
    float bv0 = b2[nt0 * 16 + m16];
    float bv1 = b2[(nt0 + 1) * 16 + m16];
#pragma unroll
    for (int i = 0; i < 8; ++i) { d0[i] = bv0; d1[i] = bv1; }
  }
#pragma unroll
  for (int kk = 0; kk < 4; ++kk) {
    v16bf a  = load_a_frag(H1, H1_STRIDE, kk, m16, hi);
    v16bf bA = load_b_frag(w2s, nt0,     4, kk, lane);
    v16bf bB = load_b_frag(w2s, nt0 + 1, 4, kk, lane);
    d0 = wmma_bf16(a, bA, d0);
    d1 = wmma_bf16(a, bB, d1);
  }
#pragma unroll
  for (int r = 0; r < 8; ++r) {
    int row = r + hi * 8;
    H2[row * H2_STRIDE + nt0 * 16 + m16]       = d0[r];
    H2[row * H2_STRIDE + (nt0 + 1) * 16 + m16] = d1[r];
  }
  __syncthreads();

  // ---- LayerNorm over 128 cols per row ----
  {
    int r = t >> 3, p = t & 7;
    float s = 0.f, q = 0.f;
#pragma unroll
    for (int i = 0; i < 16; ++i) {
      float v = H2[r * H2_STRIDE + p * 16 + i];
      s += v; q += v * v;
    }
    part[(r * 8 + p) * 2]     = s;
    part[(r * 8 + p) * 2 + 1] = q;
  }
  __syncthreads();
  if (t < 16) {
    float s = 0.f, q = 0.f;
#pragma unroll
    for (int p = 0; p < 8; ++p) { s += part[(t * 8 + p) * 2]; q += part[(t * 8 + p) * 2 + 1]; }
    float mu  = s * (1.0f / 128.0f);
    float var = q * (1.0f / 128.0f) - mu * mu;
    stats[t * 2]     = mu;
    stats[t * 2 + 1] = rsqrtf(var + LN_EPS);
  }
  __syncthreads();

  // ---- affine + residual, write edges_new, scatter-add into agg ----
  {
    const int c = t;
    const float gc = g[c], bc = beta[c];
#pragma unroll 4
    for (int m = 0; m < 16; ++m) {
      float v = (H2[m * H2_STRIDE + c] - stats[m * 2]) * stats[m * 2 + 1] * gc + bc
              + edge_attr[(size_t)(e0 + m) * DFEAT + c];
      out_edges[(size_t)(e0 + m) * DFEAT + c] = v;
      atomic_add_f32(&agg[(size_t)didx[m] * DFEAT + c], v);
    }
  }
}

// ---------------------------------------------------------------------------
// Node MLP: A = concat([a1_row, a2_row]) (K = 256), residual = a1, 16 rows/block.
// Used for dst (a1=x_dst, a2=agg) and src (a1=x_src, a2=x_src).
// ---------------------------------------------------------------------------
#define NA_STRIDE 264   // 256 + 8 pad: row = 528B = 33*16B

__global__ __launch_bounds__(128)
void node_kernel(const float* __restrict__ a1, const float* __restrict__ a2,
                 const bf16_t* __restrict__ w1s, const float* __restrict__ b1,
                 const bf16_t* __restrict__ w2s, const float* __restrict__ b2,
                 const float* __restrict__ g, const float* __restrict__ beta,
                 float* __restrict__ out) {
  __shared__ __align__(16) bf16_t Asta[16 * NA_STRIDE];
  __shared__ __align__(16) bf16_t H1[16 * H1_STRIDE];
  __shared__ __align__(16) float  H2[16 * H2_STRIDE];
  __shared__ float part[16 * 8 * 2];
  __shared__ float stats[16 * 2];

  const int t  = threadIdx.x;
  const int r0 = blockIdx.x * 16;

  {
    const int c = t;
#pragma unroll 4
    for (int m = 0; m < 16; ++m) {
      Asta[m * NA_STRIDE + c]       = (bf16_t)a1[(size_t)(r0 + m) * DFEAT + c];
      Asta[m * NA_STRIDE + 128 + c] = (bf16_t)a2[(size_t)(r0 + m) * DFEAT + c];
    }
  }
  __syncthreads();

  const int lane = t & 31;
  const int wave = t >> 5;
  const int m16  = lane & 15;
  const int hi   = lane >> 4;
  const int nt0  = wave * 2;

  v8f acc0, acc1;
  {
    float bv0 = b1[nt0 * 16 + m16];
    float bv1 = b1[(nt0 + 1) * 16 + m16];
#pragma unroll
    for (int i = 0; i < 8; ++i) { acc0[i] = bv0; acc1[i] = bv1; }
  }
#pragma unroll
  for (int kk = 0; kk < 8; ++kk) {
    v16bf a  = load_a_frag(Asta, NA_STRIDE, kk, m16, hi);
    v16bf bA = load_b_frag(w1s, nt0,     8, kk, lane);
    v16bf bB = load_b_frag(w1s, nt0 + 1, 8, kk, lane);
    acc0 = wmma_bf16(a, bA, acc0);
    acc1 = wmma_bf16(a, bB, acc1);
  }
#pragma unroll
  for (int r = 0; r < 8; ++r) {
    int row = r + hi * 8;
    H1[row * H1_STRIDE + nt0 * 16 + m16]       = (bf16_t)silu(acc0[r]);
    H1[row * H1_STRIDE + (nt0 + 1) * 16 + m16] = (bf16_t)silu(acc1[r]);
  }
  __syncthreads();

  v8f d0, d1;
  {
    float bv0 = b2[nt0 * 16 + m16];
    float bv1 = b2[(nt0 + 1) * 16 + m16];
#pragma unroll
    for (int i = 0; i < 8; ++i) { d0[i] = bv0; d1[i] = bv1; }
  }
#pragma unroll
  for (int kk = 0; kk < 4; ++kk) {
    v16bf a  = load_a_frag(H1, H1_STRIDE, kk, m16, hi);
    v16bf bA = load_b_frag(w2s, nt0,     4, kk, lane);
    v16bf bB = load_b_frag(w2s, nt0 + 1, 4, kk, lane);
    d0 = wmma_bf16(a, bA, d0);
    d1 = wmma_bf16(a, bB, d1);
  }
#pragma unroll
  for (int r = 0; r < 8; ++r) {
    int row = r + hi * 8;
    H2[row * H2_STRIDE + nt0 * 16 + m16]       = d0[r];
    H2[row * H2_STRIDE + (nt0 + 1) * 16 + m16] = d1[r];
  }
  __syncthreads();

  {
    int r = t >> 3, p = t & 7;
    float s = 0.f, q = 0.f;
#pragma unroll
    for (int i = 0; i < 16; ++i) {
      float v = H2[r * H2_STRIDE + p * 16 + i];
      s += v; q += v * v;
    }
    part[(r * 8 + p) * 2]     = s;
    part[(r * 8 + p) * 2 + 1] = q;
  }
  __syncthreads();
  if (t < 16) {
    float s = 0.f, q = 0.f;
#pragma unroll
    for (int p = 0; p < 8; ++p) { s += part[(t * 8 + p) * 2]; q += part[(t * 8 + p) * 2 + 1]; }
    float mu  = s * (1.0f / 128.0f);
    float var = q * (1.0f / 128.0f) - mu * mu;
    stats[t * 2]     = mu;
    stats[t * 2 + 1] = rsqrtf(var + LN_EPS);
  }
  __syncthreads();
  {
    const int c = t;
    const float gc = g[c], bc = beta[c];
#pragma unroll 4
    for (int m = 0; m < 16; ++m) {
      float v = (H2[m * H2_STRIDE + c] - stats[m * 2]) * stats[m * 2 + 1] * gc + bc
              + a1[(size_t)(r0 + m) * DFEAT + c];
      out[(size_t)(r0 + m) * DFEAT + c] = v;
    }
  }
}

// ---------------------------------------------------------------------------
// Launch
// ---------------------------------------------------------------------------
extern "C" void kernel_launch(void* const* d_in, const int* in_sizes, int n_in,
                              void* d_out, int out_size, void* d_ws, size_t ws_size,
                              hipStream_t stream) {
  const float* x_src     = (const float*)d_in[0];
  const float* x_dst     = (const float*)d_in[1];
  const float* edge_attr = (const float*)d_in[2];
  const int*   eidx      = (const int*)d_in[3];
  const float* ew1 = (const float*)d_in[4];
  const float* eb1 = (const float*)d_in[5];
  const float* ew2 = (const float*)d_in[6];
  const float* eb2 = (const float*)d_in[7];
  const float* eg  = (const float*)d_in[8];
  const float* ebt = (const float*)d_in[9];
  const float* nw1 = (const float*)d_in[10];
  const float* nb1 = (const float*)d_in[11];
  const float* nw2 = (const float*)d_in[12];
  const float* nb2 = (const float*)d_in[13];
  const float* ng  = (const float*)d_in[14];
  const float* nbt = (const float*)d_in[15];

  const int n_src  = in_sizes[0] / DFEAT;   // 100000
  const int n_dst  = in_sizes[1] / DFEAT;   // 40000
  const int n_edge = in_sizes[3] / 2;       // 600000

  // Workspace layout: [agg fp32 | edge_w1 bf16 | edge_w2 | node_w1 | node_w2]
  char* ws = (char*)d_ws;
  size_t off = 0;
  float* agg = (float*)(ws + off);
  off += (size_t)n_dst * DFEAT * sizeof(float);
  off = (off + 255) & ~(size_t)255;
  bf16_t* ew1s = (bf16_t*)(ws + off); off += (size_t)384 * 128 * 2; off = (off + 255) & ~(size_t)255;
  bf16_t* ew2s = (bf16_t*)(ws + off); off += (size_t)128 * 128 * 2; off = (off + 255) & ~(size_t)255;
  bf16_t* nw1s = (bf16_t*)(ws + off); off += (size_t)256 * 128 * 2; off = (off + 255) & ~(size_t)255;
  bf16_t* nw2s = (bf16_t*)(ws + off);

  float* out_src  = (float*)d_out;
  float* out_dstp = out_src + (size_t)n_src * DFEAT;
  float* out_edge = out_dstp + (size_t)n_dst * DFEAT;

  const int naggz = n_dst * DFEAT;
  zero_f32<<<(naggz + 255) / 256, 256, 0, stream>>>(agg, naggz);
  swizzle_w<<<(384 * 128 + 255) / 256, 256, 0, stream>>>(ew1, ew1s, 384);
  swizzle_w<<<(128 * 128 + 255) / 256, 256, 0, stream>>>(ew2, ew2s, 128);
  swizzle_w<<<(256 * 128 + 255) / 256, 256, 0, stream>>>(nw1, nw1s, 256);
  swizzle_w<<<(128 * 128 + 255) / 256, 256, 0, stream>>>(nw2, nw2s, 128);

  edge_kernel<<<n_edge / 16, 128, 0, stream>>>(
      x_src, x_dst, edge_attr, eidx, n_edge,
      ew1s, eb1, ew2s, eb2, eg, ebt, out_edge, agg);

  node_kernel<<<n_dst / 16, 128, 0, stream>>>(
      x_dst, agg, nw1s, nb1, nw2s, nb2, ng, nbt, out_dstp);

  node_kernel<<<n_src / 16, 128, 0, stream>>>(
      x_src, x_src, nw1s, nb1, nw2s, nb2, ng, nbt, out_src);
}
